// UnfoldedWMMSE_80625126081208
// MI455X (gfx1250) — matrix operand order repaired and checked by reading, not compiled, not run on hardware
//
#include <hip/hip_runtime.h>
#include <hip/hip_bf16.h>

// ---------------------------------------------------------------------------
// UnfoldedWMMSE on gfx1250 (MI455X).
// Strategy: split-precision f16 WMMA (hi/lo) for all big GEMMs, fused
// two-layer RNN with LDS-resident recurrent state, scalar per-batch WMMSE.
// ---------------------------------------------------------------------------

typedef __attribute__((ext_vector_type(16))) _Float16 v16h;
typedef __attribute__((ext_vector_type(8)))  _Float16 v8h;
typedef __attribute__((ext_vector_type(8)))  float    v8f;

#define HS 520          // LDS row stride (halfs) for 512-wide activation tiles
#define XS 136          // LDS row stride (halfs) for 128-wide x tiles
#define XPLANE (16*XS)  // 2176 halfs
#define HPLANE (16*HS)  // 8320 halfs

__device__ __forceinline__ v8f wmma16(v16h a, v16h b, v8f c) {
  // D = A(16x32 f16) * B(32x16 f16) + C(16x16 f32)
  return __builtin_amdgcn_wmma_f32_16x16x32_f16(
      /*neg_a=*/false, a, /*neg_b=*/false, b,
      /*c_mod=*/(short)0, c, /*reuse_a=*/false, /*reuse_b=*/false);
}

// A-fragment (16x32, f16) from an LDS plane, per ISA 7.12.2 layout:
// lane L<16: M=L, K = kb + {0..7} and kb+16+{0..7}
// lane L>=16: M=L-16, K = kb+8+{0..7} and kb+24+{0..7}
__device__ __forceinline__ v16h load_a(const _Float16* plane, int rs, int kb, int lane) {
  const int m   = lane & 15;
  const int kh2 = (lane >> 4) & 1;
  const _Float16* p = plane + m * rs + kb + kh2 * 8;
  v8h g0 = *(const v8h*)p;
  v8h g1 = *(const v8h*)(p + 16);
  v16h a;
#pragma unroll
  for (int i = 0; i < 8; ++i) { a[i] = g0[i]; a[8 + i] = g1[i]; }
  return a;
}

// B-fragment (32x16, f16) from global weights stored row-major W[N][K]:
// lane L<16: col N=nb+L, K = kb+{0..7} and kb+16+{0..7};  lane L>=16: col N=nb+L-16, K = kb+8+{0..7} and kb+24+{0..7}
// -> two 16B loads per lane.
__device__ __forceinline__ v16h load_b(const _Float16* Wp, int Kc, int nb, int kb, int lane) {
  const int col = nb + (lane & 15);
  const int kh2 = (lane >> 4) & 1;
  const _Float16* p = Wp + (size_t)col * Kc + kb + kh2 * 8;
  v8h g0 = *(const v8h*)p;
  v8h g1 = *(const v8h*)(p + 16);
  v16h b;
#pragma unroll
  for (int i = 0; i < 8; ++i) { b[i] = g0[i]; b[8 + i] = g1[i]; }
  return b;
}

// ---------------------------------------------------------------------------
// Prep kernels
// ---------------------------------------------------------------------------
__global__ void split_cat_kernel(_Float16* __restrict__ hi, _Float16* __restrict__ lo,
                                 const float* __restrict__ s0, int K0,
                                 const float* __restrict__ s1, int K1, int rows) {
  const int K = K0 + K1;
  const int total = rows * K;
  for (int e = blockIdx.x * blockDim.x + threadIdx.x; e < total;
       e += gridDim.x * blockDim.x) {
    const int n = e / K;
    const int k = e - n * K;
    const float v = (k < K0) ? s0[(size_t)n * K0 + k] : s1[(size_t)n * K1 + (k - K0)];
    const _Float16 h = (_Float16)v;
    hi[e] = h;
    lo[e] = (_Float16)(v - (float)h);
  }
}

__global__ void bias_sum_kernel(float* __restrict__ dst, const float* __restrict__ a,
                                const float* __restrict__ b, int n) {
  int i = blockIdx.x * blockDim.x + threadIdx.x;
  if (i < n) dst[i] = a[i] + b[i];
}

__global__ void copy_kernel(float* __restrict__ dst, const float* __restrict__ src, int n) {
  int i = blockIdx.x * blockDim.x + threadIdx.x;
  if (i < n) dst[i] = src[i];
}

// ---------------------------------------------------------------------------
// Fused two-layer tanh-RNN. Block = 16 batches x all 64 steps x both layers.
// Recurrent state kept in LDS as hi/lo f16 planes (ping-pong).
// ---------------------------------------------------------------------------
__global__ void __launch_bounds__(512)
rnn_fused_kernel(const float* __restrict__ x,
                 const _Float16* __restrict__ W0h, const _Float16* __restrict__ W0l,
                 const float* __restrict__ bcat0,
                 const _Float16* __restrict__ W1h, const _Float16* __restrict__ W1l,
                 const float* __restrict__ bcat1,
                 float* __restrict__ last) {
  extern __shared__ _Float16 smem[];
  _Float16* sxh = smem;                    // x_t hi
  _Float16* sxl = sxh + XPLANE;            // x_t lo
  _Float16* h0h = sxl + XPLANE;            // [2][HPLANE]
  _Float16* h0l = h0h + 2 * HPLANE;
  _Float16* h1h = h0l + 2 * HPLANE;
  _Float16* h1l = h1h + 2 * HPLANE;

  const int tid  = threadIdx.x;
  const int lane = tid & 31;
  const int w    = tid >> 5;               // 16 waves
  const int b0   = blockIdx.x * 16;

  // zero the parity-0 state planes (initial hidden state = 0)
  for (int e = tid; e < HPLANE; e += 512) {
    h0h[e] = (_Float16)0.f; h0l[e] = (_Float16)0.f;
    h1h[e] = (_Float16)0.f; h1l[e] = (_Float16)0.f;
  }
  __syncthreads();

  const int nb0  = w * 32;                 // each wave owns 2 N-tiles (32 cols)
  const int nb1  = w * 32 + 16;
  const int colL = lane & 15;
  const int kh   = lane >> 4;
  const float bias00 = bcat0[nb0 + colL];
  const float bias01 = bcat0[nb1 + colL];
  const float bias10 = bcat1[nb0 + colL];
  const float bias11 = bcat1[nb1 + colL];

  for (int t = 0; t < 64; ++t) {
    const int rp = t & 1;                  // read parity
    const int wp = rp ^ 1;                 // write parity

    // ---- stage x_t (16 x 128) as hi/lo f16 ----
    for (int e = tid; e < 16 * 128; e += 512) {
      const int m = e >> 7, i = e & 127;
      const float v = x[(((size_t)(b0 + m)) * 64 + t) * 128 + i];
      const _Float16 h = (_Float16)v;
      sxh[m * XS + i] = h;
      sxl[m * XS + i] = (_Float16)(v - (float)h);
    }
    __syncthreads();

    // ---- layer 0: [x_t | h0_{t-1}] @ [W_ih0|W_hh0]^T, K = 640 ----
    {
      v8f c0 = {};
      v8f c1 = {};
      for (int kt = 0; kt < 20; ++kt) {
        const _Float16 *ap_h, *ap_l;
        int kb, rs;
        if (kt < 4) { ap_h = sxh; ap_l = sxl; kb = kt * 32; rs = XS; }
        else        { ap_h = h0h + rp * HPLANE; ap_l = h0l + rp * HPLANE;
                      kb = (kt - 4) * 32; rs = HS; }
        const v16h Ah = load_a(ap_h, rs, kb, lane);
        const v16h Al = load_a(ap_l, rs, kb, lane);
        const int kw = kt * 32;
        const v16h Bh0 = load_b(W0h, 640, nb0, kw, lane);
        const v16h Bl0 = load_b(W0l, 640, nb0, kw, lane);
        const v16h Bh1 = load_b(W0h, 640, nb1, kw, lane);
        const v16h Bl1 = load_b(W0l, 640, nb1, kw, lane);
        c0 = wmma16(Ah, Bh0, c0);
        c0 = wmma16(Ah, Bl0, c0);
        c0 = wmma16(Al, Bh0, c0);
        c1 = wmma16(Ah, Bh1, c1);
        c1 = wmma16(Ah, Bl1, c1);
        c1 = wmma16(Al, Bh1, c1);
      }
      _Float16* dh = h0h + wp * HPLANE;
      _Float16* dl = h0l + wp * HPLANE;
#pragma unroll
      for (int r = 0; r < 8; ++r) {
        const int row = r + kh * 8;
        const float v0 = tanhf(c0[r] + bias00);
        _Float16 hv = (_Float16)v0;
        dh[row * HS + nb0 + colL] = hv;
        dl[row * HS + nb0 + colL] = (_Float16)(v0 - (float)hv);
        const float v1 = tanhf(c1[r] + bias01);
        hv = (_Float16)v1;
        dh[row * HS + nb1 + colL] = hv;
        dl[row * HS + nb1 + colL] = (_Float16)(v1 - (float)hv);
      }
    }
    __syncthreads();

    // ---- layer 1: [h0_t | h1_{t-1}] @ [W_ih1|W_hh1]^T, K = 1024 ----
    {
      v8f c0 = {};
      v8f c1 = {};
      for (int kt = 0; kt < 32; ++kt) {
        const _Float16 *ap_h, *ap_l;
        int kb;
        if (kt < 16) { ap_h = h0h + wp * HPLANE; ap_l = h0l + wp * HPLANE;
                       kb = kt * 32; }
        else         { ap_h = h1h + rp * HPLANE; ap_l = h1l + rp * HPLANE;
                       kb = (kt - 16) * 32; }
        const v16h Ah = load_a(ap_h, HS, kb, lane);
        const v16h Al = load_a(ap_l, HS, kb, lane);
        const int kw = kt * 32;
        const v16h Bh0 = load_b(W1h, 1024, nb0, kw, lane);
        const v16h Bl0 = load_b(W1l, 1024, nb0, kw, lane);
        const v16h Bh1 = load_b(W1h, 1024, nb1, kw, lane);
        const v16h Bl1 = load_b(W1l, 1024, nb1, kw, lane);
        c0 = wmma16(Ah, Bh0, c0);
        c0 = wmma16(Ah, Bl0, c0);
        c0 = wmma16(Al, Bh0, c0);
        c1 = wmma16(Ah, Bh1, c1);
        c1 = wmma16(Ah, Bl1, c1);
        c1 = wmma16(Al, Bh1, c1);
      }
      _Float16* dh = h1h + wp * HPLANE;
      _Float16* dl = h1l + wp * HPLANE;
#pragma unroll
      for (int r = 0; r < 8; ++r) {
        const int row = r + kh * 8;
        const float v0 = tanhf(c0[r] + bias10);
        _Float16 hv = (_Float16)v0;
        dh[row * HS + nb0 + colL] = hv;
        dl[row * HS + nb0 + colL] = (_Float16)(v0 - (float)hv);
        const float v1 = tanhf(c1[r] + bias11);
        hv = (_Float16)v1;
        dh[row * HS + nb1 + colL] = hv;
        dl[row * HS + nb1 + colL] = (_Float16)(v1 - (float)hv);
        if (t == 63) {
          last[((size_t)(b0 + row)) * 512 + nb0 + colL] = v0;
          last[((size_t)(b0 + row)) * 512 + nb1 + colL] = v1;
        }
      }
    }
    __syncthreads();
  }
}

// ---------------------------------------------------------------------------
// Head GEMM: ohead[2048, 1728] = last[2048,512] @ Whead[1728,512]^T + bias
// ---------------------------------------------------------------------------
__global__ void __launch_bounds__(512)
head_gemm_kernel(const float* __restrict__ lastA,
                 const _Float16* __restrict__ Wh, const _Float16* __restrict__ Wl,
                 const float* __restrict__ bias, float* __restrict__ outo) {
  __shared__ _Float16 sAh[16 * HS];
  __shared__ _Float16 sAl[16 * HS];
  const int tid = threadIdx.x, lane = tid & 31, w = tid >> 5;
  const int b0 = blockIdx.x * 16;
  for (int e = tid; e < 16 * 512; e += 512) {
    const int m = e >> 9, k = e & 511;
    const float v = lastA[((size_t)(b0 + m)) * 512 + k];
    const _Float16 h = (_Float16)v;
    sAh[m * HS + k] = h;
    sAl[m * HS + k] = (_Float16)(v - (float)h);
  }
  __syncthreads();
  const int colL = lane & 15, kh = lane >> 4;
  for (int nt = w; nt < 108; nt += 16) {       // uniform per wave
    const int nb = nt * 16;
    v8f c = {};
    for (int kt = 0; kt < 16; ++kt) {
      const v16h Ah = load_a(sAh, HS, kt * 32, lane);
      const v16h Al = load_a(sAl, HS, kt * 32, lane);
      const v16h Bh = load_b(Wh, 512, nb, kt * 32, lane);
      const v16h Bl = load_b(Wl, 512, nb, kt * 32, lane);
      c = wmma16(Ah, Bh, c);
      c = wmma16(Ah, Bl, c);
      c = wmma16(Al, Bh, c);
    }
    const int col = nb + colL;
    const float bb = bias[col];
#pragma unroll
    for (int r = 0; r < 8; ++r) {
      const int row = r + kh * 8;
      outo[((size_t)(b0 + row)) * 1728 + col] = c[r] + bb;
    }
  }
}

// ---------------------------------------------------------------------------
// WMMSE + PGD, one thread per batch. ohead row = [o_irs(128) | o_ch(1600)].
// out row (192) = [out_bs(64) | tre(64) | tim(64)]
// ---------------------------------------------------------------------------
__global__ void wmmse_kernel(const float* __restrict__ ohead,
                             const float* __restrict__ lr,
                             const float* __restrict__ bfinit,   // interleaved re,im
                             float* __restrict__ out) {
  const int b = blockIdx.x * blockDim.x + threadIdx.x;
  if (b >= 2048) return;
  const float* oh  = ohead + (size_t)b * 1728;
  const float* och = oh + 128;           // 1600 values: G(1024) D(64) R(512)
  float* ob = out + (size_t)b * 192;
  const float s2 = 0.01f;
  const float sqrtP = 3.1622776601683795f;

  // Hpre[a][u] = D[a][u] + sum_r G[a][r]*theta[r]*R[r][u]
  float Hr[8][4], Hi[8][4];
#pragma unroll
  for (int a = 0; a < 8; ++a)
#pragma unroll
    for (int u = 0; u < 4; ++u) {
      Hr[a][u] = och[1024 + a * 4 + u];
      Hi[a][u] = och[1056 + a * 4 + u];
    }
  for (int r = 0; r < 64; ++r) {
    const float re = oh[r], im = oh[64 + r];
    const float f = sqrtf(re * re + im * im);
    const float thr = re / f, thi = im / f;
    ob[64 + r]  = thr;                   // norm_irs real part
    ob[128 + r] = thi;                   // norm_irs imag part
    float Rr[4], Ri[4];
#pragma unroll
    for (int u = 0; u < 4; ++u) {
      Rr[u] = och[1088 + r * 4 + u];
      Ri[u] = och[1344 + r * 4 + u];
    }
#pragma unroll
    for (int a = 0; a < 8; ++a) {
      const float Gre = och[a * 64 + r], Gim = och[512 + a * 64 + r];
      const float gtr = Gre * thr - Gim * thi;
      const float gti = Gre * thi + Gim * thr;
#pragma unroll
      for (int u = 0; u < 4; ++u) {
        Hr[a][u] += gtr * Rr[u] - gti * Ri[u];
        Hi[a][u] += gtr * Ri[u] + gti * Rr[u];
      }
    }
  }

  float Br[8][4], Bi[8][4];
#pragma unroll
  for (int a = 0; a < 8; ++a)
#pragma unroll
    for (int u = 0; u < 4; ++u) {
      Br[a][u] = bfinit[2 * (a * 4 + u)];
      Bi[a][u] = bfinit[2 * (a * 4 + u) + 1];
    }

  for (int i = 0; i < 5; ++i) {
    // prod[u][v] = sum_a conj(Hm[u][a]) * BF[a][v];  Hm[u][a] = Hpre[a][u]
    float pr[4][4], pi[4][4];
#pragma unroll
    for (int u = 0; u < 4; ++u)
#pragma unroll
      for (int v = 0; v < 4; ++v) {
        float sr = 0.f, si = 0.f;
#pragma unroll
        for (int a = 0; a < 8; ++a) {
          sr += Hr[a][u] * Br[a][v] + Hi[a][u] * Bi[a][v];
          si += Hr[a][u] * Bi[a][v] - Hi[a][u] * Br[a][v];
        }
        pr[u][v] = sr; pi[u][v] = si;
      }
    float wr_[4], UWr[4], UWi[4];
#pragma unroll
    for (int u = 0; u < 4; ++u) {
      float sum1 = 0.f;
#pragma unroll
      for (int v = 0; v < 4; ++v) sum1 += pr[u][v] * pr[u][v] + pi[u][v] * pi[u][v];
      const float pasd = pr[u][u] * pr[u][u] + pi[u][u] * pi[u][u];
      const float den = sum1 + s2;
      const float Uir = pr[u][u] / den, Uii = pi[u][u] / den;
      const float Wi_ = den / (sum1 - pasd + s2);
      wr_[u] = (Uir * Uir + Uii * Uii) * Wi_;
      UWr[u] = Uir * Wi_;
      UWi[u] = Uii * Wi_;
    }
    // Amat[a][c] = sum_u wr[u] * Hpre[a][u] * conj(Hpre[c][u]);  bvec = Hpre .* (Ui*Wi)
    float Ar[8][8], Ai[8][8], bvr[8][4], bvi[8][4];
#pragma unroll
    for (int a = 0; a < 8; ++a) {
#pragma unroll
      for (int c = 0; c < 8; ++c) {
        float sr = 0.f, si = 0.f;
#pragma unroll
        for (int u = 0; u < 4; ++u) {
          sr += wr_[u] * (Hr[a][u] * Hr[c][u] + Hi[a][u] * Hi[c][u]);
          si += wr_[u] * (Hi[a][u] * Hr[c][u] - Hr[a][u] * Hi[c][u]);
        }
        Ar[a][c] = sr; Ai[a][c] = si;
      }
#pragma unroll
      for (int u = 0; u < 4; ++u) {
        bvr[a][u] = Hr[a][u] * UWr[u] - Hi[a][u] * UWi[u];
        bvi[a][u] = Hr[a][u] * UWi[u] + Hi[a][u] * UWr[u];
      }
    }
    for (int j = 0; j < 5; ++j) {
      const float lrv = lr[i * 5 + j];
      float gr[8][4], gi[8][4];
#pragma unroll
      for (int a = 0; a < 8; ++a)
#pragma unroll
        for (int v = 0; v < 4; ++v) {
          float sr = -bvr[a][v], si = -bvi[a][v];
#pragma unroll
          for (int c = 0; c < 8; ++c) {
            sr += Ar[a][c] * Br[c][v] - Ai[a][c] * Bi[c][v];
            si += Ar[a][c] * Bi[c][v] + Ai[a][c] * Br[c][v];
          }
          gr[a][v] = 2.f * sr; gi[a][v] = 2.f * si;
        }
      float nsq = 0.f;
#pragma unroll
      for (int a = 0; a < 8; ++a)
#pragma unroll
        for (int v = 0; v < 4; ++v) {
          Br[a][v] -= lrv * gr[a][v];
          Bi[a][v] -= lrv * gi[a][v];
          nsq += Br[a][v] * Br[a][v] + Bi[a][v] * Bi[a][v];
        }
      const float nrm = sqrtf(nsq);
      float ex = nrm - sqrtP;
      if (ex < 0.f) ex = 0.f;
      const float sc = sqrtP / (sqrtP + ex);
#pragma unroll
      for (int a = 0; a < 8; ++a)
#pragma unroll
        for (int v = 0; v < 4; ++v) { Br[a][v] *= sc; Bi[a][v] *= sc; }
    }
  }
  // out_bs: BF.swapaxes(1,2).reshape(32) -> idx = u*8 + a; [real | imag]
#pragma unroll
  for (int u = 0; u < 4; ++u)
#pragma unroll
    for (int a = 0; a < 8; ++a) {
      ob[u * 8 + a]      = Br[a][u];
      ob[32 + u * 8 + a] = Bi[a][u];
    }
}

// ---------------------------------------------------------------------------
// Launch
// ---------------------------------------------------------------------------
extern "C" void kernel_launch(void* const* d_in, const int* in_sizes, int n_in,
                              void* d_out, int out_size, void* d_ws, size_t ws_size,
                              hipStream_t stream) {
  (void)in_sizes; (void)n_in; (void)out_size; (void)ws_size;
  const float* x     = (const float*)d_in[0];
  const float* W_ih0 = (const float*)d_in[1];
  const float* W_hh0 = (const float*)d_in[2];
  const float* b_ih0 = (const float*)d_in[3];
  const float* b_hh0 = (const float*)d_in[4];
  const float* W_ih1 = (const float*)d_in[5];
  const float* W_hh1 = (const float*)d_in[6];
  const float* b_ih1 = (const float*)d_in[7];
  const float* b_hh1 = (const float*)d_in[8];
  const float* W_irs = (const float*)d_in[9];
  const float* b_irs = (const float*)d_in[10];
  const float* W_ch  = (const float*)d_in[11];
  const float* b_ch  = (const float*)d_in[12];
  const float* lr    = (const float*)d_in[13];
  const float* bfini = (const float*)d_in[14];   // complex64 -> interleaved floats
  float* out = (float*)d_out;

  char* ws = (char*)d_ws;
  size_t off = 0;
  auto carve = [&](size_t bytes) -> void* {
    off = (off + 255) & ~(size_t)255;
    void* p = ws + off;
    off += bytes;
    return p;
  };
  _Float16* W0h = (_Float16*)carve((size_t)512 * 640 * 2);
  _Float16* W0l = (_Float16*)carve((size_t)512 * 640 * 2);
  _Float16* W1h = (_Float16*)carve((size_t)512 * 1024 * 2);
  _Float16* W1l = (_Float16*)carve((size_t)512 * 1024 * 2);
  _Float16* WHh = (_Float16*)carve((size_t)1728 * 512 * 2);
  _Float16* WHl = (_Float16*)carve((size_t)1728 * 512 * 2);
  float* bcat0 = (float*)carve(512 * 4);
  float* bcat1 = (float*)carve(512 * 4);
  float* bhead = (float*)carve(1728 * 4);
  float* last  = (float*)carve((size_t)2048 * 512 * 4);
  float* ohead = (float*)carve((size_t)2048 * 1728 * 4);

  // weight prep
  split_cat_kernel<<<320, 256, 0, stream>>>(W0h, W0l, W_ih0, 128, W_hh0, 512, 512);
  split_cat_kernel<<<512, 256, 0, stream>>>(W1h, W1l, W_ih1, 512, W_hh1, 512, 512);
  split_cat_kernel<<<64,  256, 0, stream>>>(WHh, WHl, W_irs, 512, nullptr, 0, 128);
  split_cat_kernel<<<800, 256, 0, stream>>>(WHh + (size_t)128 * 512,
                                            WHl + (size_t)128 * 512,
                                            W_ch, 512, nullptr, 0, 1600);
  bias_sum_kernel<<<2, 256, 0, stream>>>(bcat0, b_ih0, b_hh0, 512);
  bias_sum_kernel<<<2, 256, 0, stream>>>(bcat1, b_ih1, b_hh1, 512);
  copy_kernel<<<1, 128, 0, stream>>>(bhead, b_irs, 128);
  copy_kernel<<<7, 256, 0, stream>>>(bhead + 128, b_ch, 1600);

  // fused 2-layer RNN: 128 blocks x 16 batches, dynamic LDS 141824 B
  const size_t smem = (size_t)(2 * XPLANE + 8 * HPLANE) * sizeof(_Float16);
  rnn_fused_kernel<<<128, 512, smem, stream>>>(x, W0h, W0l, bcat0,
                                               W1h, W1l, bcat1, last);

  // heads
  head_gemm_kernel<<<128, 512, 0, stream>>>(last, WHh, WHl, bhead, ohead);

  // WMMSE + output assembly
  wmmse_kernel<<<8, 256, 0, stream>>>(ohead, lr, bfini, out);
}
